// TransformerBlockLite_80298708566523
// MI455X (gfx1250) — compile-verified, matrix operand order checked by reading, not run on hardware
//
#include <hip/hip_runtime.h>

// ---------------------------------------------------------------------------
// TransformerBlockLite for MI455X (gfx1250): all 1x1-conv GEMMs + linear
// attention on v_wmma_f32_16x16x32_f16 (wave32). f16 intermediates, fused
// epilogues, residuals accumulated into d_out. Activation tiles staged to LDS
// via GLOBAL_LOAD_ASYNC_TO_LDS_B128 (asynccnt-tracked DMA path).
// ---------------------------------------------------------------------------

typedef __attribute__((ext_vector_type(16))) _Float16 v16h;
typedef __attribute__((ext_vector_type(8)))  _Float16 v8h;
typedef __attribute__((ext_vector_type(8)))  float    v8f;
typedef __attribute__((ext_vector_type(4)))  int      v4i;

#define BB    8
#define CC    256
#define HH    128
#define WW    128
#define NN    (HH*WW)          // 16384
#define HEADS 8
#define HDIM  32
#define INNER 256
#define HID   512

#if defined(__HIP_DEVICE_COMPILE__) && __has_builtin(__builtin_amdgcn_global_load_async_to_lds_b128)
#define ASYNC_LDS 1
#endif

// Fragment loader: per the CDNA5 16-bit A/B layout, a lane holds K = kb..kb+7
// and K = kb+16..kb+23 (kb = (lane>=16)?8:0), contiguous in memory.
__device__ __forceinline__ v16h load_frag(const _Float16* p) {
    v8h lo = *(const v8h*)(p);
    v8h hi = *(const v8h*)(p + 16);
    return __builtin_shufflevector(lo, hi, 0,1,2,3,4,5,6,7,8,9,10,11,12,13,14,15);
}

__device__ __forceinline__ v8f wmma_f16(v16h a, v16h b, v8f c) {
    return __builtin_amdgcn_wmma_f32_16x16x32_f16(false, a, false, b, (short)0, c, false, false);
}

// ---------------------------------------------------------------------------
// f32 -> f16 cast (weights)
// ---------------------------------------------------------------------------
__global__ void cast_f16_kernel(const float* __restrict__ s, _Float16* __restrict__ d, int n) {
    int i = blockIdx.x * blockDim.x + threadIdx.x;
    if (i < n) d[i] = (_Float16)s[i];
}

// ---------------------------------------------------------------------------
// Per-pixel LayerNorm over 256 channels. In: (B,C,N) f32. Out: (B,N,C) f16
// (pixel-major so GEMM B-fragments read contiguous K=channel).
// 256 threads = 16 pixels x 16 channel-groups; x reads coalesced over n.
// ---------------------------------------------------------------------------
__global__ void ln_kernel(const float* __restrict__ x, const float* __restrict__ g,
                          const float* __restrict__ bet, _Float16* __restrict__ out) {
    __shared__ float tile[CC][16];
    __shared__ float psum[16][16];
    __shared__ float psq[16][16];
    __shared__ float smu[16];
    __shared__ float srs[16];
    const int b  = blockIdx.y;
    const int n0 = blockIdx.x * 16;
    const int px = threadIdx.x & 15;
    const int cg = threadIdx.x >> 4;
    float s = 0.f, sq = 0.f;
    for (int cc = 0; cc < 16; ++cc) {
        int c = cg * 16 + cc;
        float v = x[((size_t)(b * CC + c)) * NN + n0 + px];
        tile[c][px] = v;
        s += v; sq += v * v;
    }
    psum[cg][px] = s; psq[cg][px] = sq;
    __syncthreads();
    if (cg == 0) {
        float ts = 0.f, tq = 0.f;
        for (int i = 0; i < 16; ++i) { ts += psum[i][px]; tq += psq[i][px]; }
        float mu  = ts * (1.f / CC);
        float var = tq * (1.f / CC) - mu * mu;
        smu[px] = mu;
        srs[px] = rsqrtf(var + 1e-6f);
    }
    __syncthreads();
    const float mu = smu[px], rs = srs[px];
    _Float16* op = out + ((size_t)(b * NN + n0 + px)) * CC + cg * 16;
    for (int cc = 0; cc < 16; ++cc) {
        int c = cg * 16 + cc;
        op[cc] = (_Float16)((tile[c][px] - mu) * rs * g[c] + bet[c]);
    }
}

// ---------------------------------------------------------------------------
// Epilogues for the generic GEMM
// ---------------------------------------------------------------------------
enum { EPI_Q = 0, EPI_K, EPI_V, EPI_WO, EPI_W1, EPI_W2 };

template<int EPI>
__device__ __forceinline__ void epilogue(const v8f& acc, int b, int o0, int n0, int m0, int col,
                                         int O, _Float16* __restrict__ out_h,
                                         _Float16* __restrict__ out_h2, float* __restrict__ out_f,
                                         const float* __restrict__ bias,
                                         const float* __restrict__ resid) {
    if constexpr (EPI == EPI_K || EPI == EPI_V) {
        // store (B, O, N) f16; K gets phi = elu+1
        for (int i = 0; i < 8; ++i) {
            int o = o0 + m0 + i;
            float d = acc[i];
            if constexpr (EPI == EPI_K) d = (d > 0.f) ? (d + 1.f) : __expf(d);
            out_h[((size_t)(b * O + o)) * NN + n0 + col] = (_Float16)d;
        }
    } else if constexpr (EPI == EPI_Q) {
        // phi, store transposed (B, N, INNER) f16 -> vectorized 16B store
        v8h vs;
        for (int i = 0; i < 8; ++i) {
            float d = acc[i];
            vs[i] = (_Float16)((d > 0.f) ? (d + 1.f) : __expf(d));
        }
        *(v8h*)(out_h + ((size_t)(b * NN + n0 + col)) * O + o0 + m0) = vs;
    } else if constexpr (EPI == EPI_WO || EPI == EPI_W2) {
        // f32 out with bias + residual (WO: resid=x, W2: resid=out buffer)
        for (int i = 0; i < 8; ++i) {
            int o = o0 + m0 + i;
            size_t idx = ((size_t)(b * O + o)) * NN + n0 + col;
            out_f[idx] = acc[i] + bias[o] + resid[idx];
        }
    } else if constexpr (EPI == EPI_W1) {
        // split: o<HID -> exact gelu -> y1g (B,HID,N); else raw gate -> y2 (B,HID,N)
        for (int i = 0; i < 8; ++i) {
            int o = o0 + m0 + i;
            float d = acc[i] + bias[o];
            if (o < HID) {
                float gl = 0.5f * d * (1.f + erff(d * 0.70710678118f));
                out_h[((size_t)(b * HID + o)) * NN + n0 + col] = (_Float16)gl;
            } else {
                out_h2[((size_t)(b * HID + (o - HID))) * NN + n0 + col] = (_Float16)d;
            }
        }
    }
}

// ---------------------------------------------------------------------------
// Generic WMMA GEMM: D = W(OxC) * X(B,N,C pixel-major)^T.
// Block = 256 threads (8 waves) sharing ONE 16-pixel tile; the contiguous
// 16 x Cdim f16 X-tile is staged into LDS with async-to-LDS loads, then each
// wave computes TWO 16x16 O-tiles, reusing the LDS B-fragment per K-step.
// A block covers 256 output channels -> per K-step: 2 ds_load + 4 global
// loads feed 2 WMMAs.
// ---------------------------------------------------------------------------
template<int EPI>
__global__ __launch_bounds__(256)
void gemm_kernel(const _Float16* __restrict__ Wm, const _Float16* __restrict__ X,
                 int O, int Cdim,
                 _Float16* __restrict__ out_h, _Float16* __restrict__ out_h2,
                 float* __restrict__ out_f,
                 const float* __restrict__ bias, const float* __restrict__ resid) {
    __shared__ _Float16 xtile[16 * 512];     // up to 16 KB (Cdim <= 512)
    const int b  = blockIdx.z;
    const int n0 = blockIdx.x * 16;
    // Stage X tile: pixels n0..n0+15 are adjacent rows in (B,N,C) -> one
    // contiguous 16*Cdim*2-byte region.
    const _Float16* xsrc = X + ((size_t)(b * NN + n0)) * Cdim;
    const int chunks = (16 * Cdim) >> 3;     // 16-byte chunks
    for (int c = threadIdx.x; c < chunks; c += 256) {
#ifdef ASYNC_LDS
        __builtin_amdgcn_global_load_async_to_lds_b128(
            (__attribute__((address_space(1))) v4i*)(xsrc + c * 8),
            (__attribute__((address_space(3))) v4i*)(xtile + c * 8), 0, 0);
#else
        *(v8h*)(xtile + c * 8) = *(const v8h*)(xsrc + c * 8);
#endif
    }
#ifdef ASYNC_LDS
#if __has_builtin(__builtin_amdgcn_s_wait_asynccnt)
    __builtin_amdgcn_s_wait_asynccnt(0);
#else
    asm volatile("s_wait_asynccnt 0" ::: "memory");
#endif
#endif
    __syncthreads();

    const int lane = threadIdx.x & 31;
    const int wave = threadIdx.x >> 5;
    const int ot0  = blockIdx.y * 16 + wave * 2;     // two o-tiles per wave
    const int r    = lane & 15;                      // A row / B col / D col
    const int hi   = lane >> 4;                      // lane half -> K selector
    const _Float16* wp0 = Wm + (size_t)(ot0 * 16 + r) * Cdim + hi * 8;
    const _Float16* wp1 = wp0 + (size_t)16 * Cdim;
    const _Float16* bp  = xtile + r * Cdim + hi * 8; // LDS B-fragment base
    v8f acc0 = {}, acc1 = {};
    for (int k = 0; k < Cdim; k += 32) {
        v16h bm = load_frag(bp + k);                 // ds_load_b128 x2
        v16h a0 = load_frag(wp0 + k);
        v16h a1 = load_frag(wp1 + k);
        acc0 = wmma_f16(a0, bm, acc0);
        acc1 = wmma_f16(a1, bm, acc1);
    }
    const int m0 = hi * 8, col = r;
    epilogue<EPI>(acc0, b, ot0 * 16,      n0, m0, col, O, out_h, out_h2, out_f, bias, resid);
    epilogue<EPI>(acc1, b, ot0 * 16 + 16, n0, m0, col, O, out_h, out_h2, out_f, bias, resid);
}

// ---------------------------------------------------------------------------
// KV[b,h,d,e] = sum_n k[d,n] * v[e,n]  (32x32 per head) via WMMA over n-chunks.
// Block = 4 waves -> the four (d-tile, e-tile) combos; f32 atomic reduction.
// ---------------------------------------------------------------------------
__global__ __launch_bounds__(128)
void kv_kernel(const _Float16* __restrict__ K16, const _Float16* __restrict__ V16,
               float* __restrict__ KV) {
    const int lane = threadIdx.x & 31;
    const int wave = threadIdx.x >> 5;
    const int mi = wave >> 1, ei = wave & 1;
    const int bh = blockIdx.y;
    const int b = bh >> 3, h = bh & 7;
    const int chunk = blockIdx.x;       // 256 n-values per chunk
    const int r = lane & 15, hi = lane >> 4;
    const _Float16* kp = K16 + ((size_t)(b * INNER + h * HDIM + mi * 16 + r)) * NN + chunk * 256 + hi * 8;
    const _Float16* vp = V16 + ((size_t)(b * INNER + h * HDIM + ei * 16 + r)) * NN + chunk * 256 + hi * 8;
    v8f acc = {};
    for (int t = 0; t < 8; ++t) {
        v16h a  = load_frag(kp + t * 32);
        v16h bm = load_frag(vp + t * 32);
        acc = wmma_f16(a, bm, acc);
    }
    for (int i = 0; i < 8; ++i) {
        int d = mi * 16 + hi * 8 + i;
        int e = ei * 16 + r;
        atomicAdd(&KV[(size_t)bh * 1024 + d * 32 + e], acc[i]);
    }
}

// ---------------------------------------------------------------------------
// out[n,e] = sum_d q[n,d] * KV[d,e] : single K=32 WMMA per 16-pixel tile.
// A = qT fragment (pixel rows), B = KV (f32->f16 on the fly, L0-resident).
// Output transposed (B, N, INNER) for the Wo GEMM.
// ---------------------------------------------------------------------------
__global__ __launch_bounds__(128)
void attn_kernel(const _Float16* __restrict__ qT, const float* __restrict__ KV,
                 _Float16* __restrict__ outT) {
    const int lane = threadIdx.x & 31;
    const int wave = threadIdx.x >> 5;
    const int combo = blockIdx.y * 4 + wave;   // 16 combos = 8 heads x 2 e-tiles
    const int h = combo >> 1, et = combo & 1;
    const int b = blockIdx.z;
    const int p0 = blockIdx.x * 16;
    const int r = lane & 15, hi = lane >> 4;
    v16h a = load_frag(qT + ((size_t)(b * NN + p0 + r)) * INNER + h * HDIM + hi * 8);
    const float* kvp = KV + ((size_t)(b * HEADS + h)) * 1024;
    const int e = et * 16 + r;
    v16h bm;
    for (int i = 0; i < 16; ++i) {
        int d = hi * 8 + ((i < 8) ? i : (8 + i));   // K striping 0-7 / 16-23
        bm[i] = (_Float16)kvp[d * 32 + e];
    }
    v8f acc = {};
    acc = wmma_f16(a, bm, acc);
    for (int i = 0; i < 8; ++i) {
        int p = p0 + hi * 8 + i;
        outT[((size_t)(b * NN + p)) * INNER + h * HDIM + et * 16 + r] = (_Float16)acc[i];
    }
}

// ---------------------------------------------------------------------------
// Depthwise 3x3 (SAME) on gelu'd y1, + bdw, * sigmoid(y2); store transposed
// (B, N, HID) f16 for the W2 GEMM.
// ---------------------------------------------------------------------------
__global__ void dwconv_kernel(const _Float16* __restrict__ y1, const _Float16* __restrict__ y2raw,
                              const float* __restrict__ wdw, const float* __restrict__ bdw,
                              _Float16* __restrict__ outT) {
    const size_t total = (size_t)BB * HID * NN;
    for (size_t idx = (size_t)blockIdx.x * blockDim.x + threadIdx.x; idx < total;
         idx += (size_t)gridDim.x * blockDim.x) {
        int n  = (int)(idx & (NN - 1));
        int ch = (int)((idx >> 14) & (HID - 1));
        int b  = (int)(idx >> 23);
        int yy = n >> 7, xx = n & (WW - 1);
        const _Float16* base = y1 + ((size_t)(b * HID + ch) << 14);
        const float* w9 = wdw + ch * 9;
        float acc = 0.f;
        #pragma unroll
        for (int dy = -1; dy <= 1; ++dy) {
            int ty = yy + dy;
            if (ty < 0 || ty >= HH) continue;
            #pragma unroll
            for (int dx = -1; dx <= 1; ++dx) {
                int tx = xx + dx;
                if (tx < 0 || tx >= WW) continue;
                acc += w9[(dy + 1) * 3 + (dx + 1)] * (float)base[ty * WW + tx];
            }
        }
        acc += bdw[ch];
        float gv = (float)y2raw[((size_t)(b * HID + ch) << 14) + n];
        float sg = 1.f / (1.f + __expf(-gv));
        outT[((size_t)(b * NN + n)) * HID + ch] = (_Float16)(acc * sg);
    }
}

// ---------------------------------------------------------------------------
// Launch. Workspace map (bytes), R = B*N*256*2 = 64 MiB:
//   [0, 1.25MB)   f16 weights | [2MB, 2.25MB) KV f32
//   R1 @ 4MB : xn16 -> outT ; R2 : k16 -> yn16 ; R3(+R4, 2R) : v16,qT -> y1g
//   R5 @ 4MB+5R : y2raw (2R) ; yactT (2R) aliases R1+R2. Peak ~ 452 MB.
// d_out doubles as the post-attention residual x1 before the final add.
// ---------------------------------------------------------------------------
extern "C" void kernel_launch(void* const* d_in, const int* in_sizes, int n_in,
                              void* d_out, int out_size, void* d_ws, size_t ws_size,
                              hipStream_t stream) {
    (void)in_sizes; (void)n_in; (void)out_size; (void)ws_size;
    const float* x    = (const float*)d_in[0];
    const float* n1g  = (const float*)d_in[1];
    const float* n1b  = (const float*)d_in[2];
    const float* wq   = (const float*)d_in[3];
    const float* wk   = (const float*)d_in[4];
    const float* wv   = (const float*)d_in[5];
    const float* wo   = (const float*)d_in[6];
    const float* bo   = (const float*)d_in[7];
    const float* n2g  = (const float*)d_in[8];
    const float* n2b  = (const float*)d_in[9];
    const float* w1   = (const float*)d_in[10];
    const float* b1   = (const float*)d_in[11];
    const float* wdw  = (const float*)d_in[12];
    const float* bdw  = (const float*)d_in[13];
    const float* w2   = (const float*)d_in[14];
    const float* b2   = (const float*)d_in[15];
    float* out = (float*)d_out;

    char* ws = (char*)d_ws;
    _Float16* wq16 = (_Float16*)ws;
    _Float16* wk16 = wq16 + 65536;
    _Float16* wv16 = wk16 + 65536;
    _Float16* wo16 = wv16 + 65536;
    _Float16* w116 = wo16 + 65536;
    _Float16* w216 = w116 + 262144;
    float*    KV   = (float*)(ws + (2u << 20));
    const size_t R = (size_t)BB * NN * 256 * 2;
    char* R1 = ws + (4u << 20);
    char* R2 = R1 + R;
    char* R3 = R2 + R;
    char* R5 = R3 + 3 * R;                  // y2raw (2R)

    _Float16* xn16  = (_Float16*)R1;
    _Float16* k16   = (_Float16*)R2;
    _Float16* v16   = (_Float16*)R3;
    _Float16* qT    = (_Float16*)(R3 + R);
    _Float16* outT  = (_Float16*)R1;        // aliases xn16 (dead by then)
    _Float16* yn16  = (_Float16*)R2;        // aliases k16
    _Float16* y1g   = (_Float16*)R3;        // 2R, aliases v16 + qT
    _Float16* y2raw = (_Float16*)R5;        // 2R
    _Float16* yactT = (_Float16*)R1;        // 2R, aliases outT + yn16

    // 1) weights -> f16
    cast_f16_kernel<<<(65536 + 255) / 256, 256, 0, stream>>>(wq, wq16, 65536);
    cast_f16_kernel<<<(65536 + 255) / 256, 256, 0, stream>>>(wk, wk16, 65536);
    cast_f16_kernel<<<(65536 + 255) / 256, 256, 0, stream>>>(wv, wv16, 65536);
    cast_f16_kernel<<<(65536 + 255) / 256, 256, 0, stream>>>(wo, wo16, 65536);
    cast_f16_kernel<<<(262144 + 255) / 256, 256, 0, stream>>>(w1, w116, 262144);
    cast_f16_kernel<<<(131072 + 255) / 256, 256, 0, stream>>>(w2, w216, 131072);

    // 2) LN1 -> xn16 (B,N,C)
    ln_kernel<<<dim3(NN / 16, BB), 256, 0, stream>>>(x, n1g, n1b, xn16);

    // 3) Q/K/V GEMMs (WMMA, LDS-staged activations)
    gemm_kernel<EPI_Q><<<dim3(NN / 16, 1, BB), 256, 0, stream>>>(wq16, xn16, INNER, CC, qT, nullptr, nullptr, nullptr, nullptr);
    gemm_kernel<EPI_K><<<dim3(NN / 16, 1, BB), 256, 0, stream>>>(wk16, xn16, INNER, CC, k16, nullptr, nullptr, nullptr, nullptr);
    gemm_kernel<EPI_V><<<dim3(NN / 16, 1, BB), 256, 0, stream>>>(wv16, xn16, INNER, CC, v16, nullptr, nullptr, nullptr, nullptr);

    // 4) KV = K V^T (WMMA + f32 atomics)
    (void)hipMemsetAsync(KV, 0, (size_t)BB * HEADS * HDIM * HDIM * sizeof(float), stream);
    kv_kernel<<<dim3(NN / 256, BB * HEADS), 128, 0, stream>>>(k16, v16, KV);

    // 5) out = Q^T KV  (single-K WMMA), transposed store
    attn_kernel<<<dim3(NN / 16, 4, BB), 128, 0, stream>>>(qT, KV, outT);

    // 6) x1 = x + Wo*out + bo  -> d_out (f32)
    gemm_kernel<EPI_WO><<<dim3(NN / 16, 1, BB), 256, 0, stream>>>(wo16, outT, CC, INNER, nullptr, nullptr, out, bo, x);

    // 7) LN2 on x1 -> yn16
    ln_kernel<<<dim3(NN / 16, BB), 256, 0, stream>>>(out, n2g, n2b, yn16);

    // 8) W1 GEMM with split gelu / gate epilogue
    gemm_kernel<EPI_W1><<<dim3(NN / 16, 4, BB), 256, 0, stream>>>(w116, yn16, 2 * HID, CC, y1g, y2raw, nullptr, b1, nullptr);

    // 9) depthwise 3x3 + bias, * sigmoid(gate), transposed store
    dwconv_kernel<<<32768, 256, 0, stream>>>(y1g, y2raw, wdw, bdw, yactT);

    // 10) final: d_out = x1 + W2*y + b2
    gemm_kernel<EPI_W2><<<dim3(NN / 16, 1, BB), 256, 0, stream>>>(w216, yactT, CC, HID, nullptr, nullptr, out, b2, out);
}
